// HGTLayer_14809047237195
// MI455X (gfx1250) — matrix operand order repaired and checked by reading, not compiled, not run on hardware
//
#include <hip/hip_runtime.h>
#include <hip/hip_bf16.h>

// ---------------------------------------------------------------------------
// HGT layer for gfx1250 (MI455X). wave32, WMMA f32_16x16x32_f16, async
// global->LDS staging of the shared B-fragment tables (ASYNCcnt path).
// Constants from the reference: IN=64, H=4, D=16, NT=2, ET=2 (H*D == IN == 64).
// ---------------------------------------------------------------------------

typedef __attribute__((ext_vector_type(16))) _Float16 v16h;
typedef __attribute__((ext_vector_type(8)))  float    v8f;

#define WMMA_F16(a, b, c) \
  __builtin_amdgcn_wmma_f32_16x16x32_f16(false, (a), false, (b), (short)0, (c), false, false)

// Async copy of 16 bytes from global to LDS. VDST holds the LDS byte address
// (flat LDS pointers truncate to the LDS offset in their low 32 bits).
__device__ __forceinline__ void async_copy_b128(uint32_t lds_addr, uint64_t gaddr) {
  asm volatile("global_load_async_to_lds_b128 %0, %1, off"
               :: "v"(lds_addr), "v"(gaddr) : "memory");
}
__device__ __forceinline__ void async_wait0() {
  asm volatile("s_wait_asynccnt 0x0" ::: "memory");
}

// ---------------------------------------------------------------------------
// Prep: fold Ratt/Rmsg into Wk/Wv, build combined B matrix per node-type
// (cols: [0:64)=Wq | [64:128)=Wk@Ratt(te0) | [128:192)=Wk@Ratt(te1)
//        | [192:256)=Wv@Rmsg(te0) | [256:320)=Wv@Rmsg(te1) )
// then pack f16 B fragments in the wave32 WMMA B layout
// (lane = column, lanes 0-15 hold K 0..7,16..23, lanes 16-31 hold K 8..15,24..31).
// ---------------------------------------------------------------------------
__global__ void __launch_bounds__(256) hgt_prep(
    const float* __restrict__ Wk, const float* __restrict__ Wq,
    const float* __restrict__ Wv, const float* __restrict__ Wa,
    const float* __restrict__ Ratt, const float* __restrict__ Rmsg,
    float* __restrict__ Bmat, _Float16* __restrict__ fragQKV,
    _Float16* __restrict__ fragWa)
{
  const int tid = threadIdx.x;
  // Phase 1: combined weight matrix, 2 types x 64 rows x 320 cols
  for (int idx = tid; idx < 2 * 64 * 320; idx += 256) {
    int t   = idx / (64 * 320);
    int rem = idx % (64 * 320);
    int i   = rem / 320;
    int c   = rem % 320;
    float v;
    if (c < 64) {
      v = Wq[(t * 64 + i) * 64 + c];
    } else {
      int cc   = c - 64;
      int kind = cc >> 7;        // 0 = katt (Wk@Ratt), 1 = vmsg (Wv@Rmsg)
      int te   = (cc >> 6) & 1;  // edge type
      int w    = cc & 63;
      int h    = w >> 4;
      int f    = w & 15;
      const float* W = kind ? Wv : Wk;
      const float* R = kind ? Rmsg : Ratt;
      float acc = 0.f;
      #pragma unroll
      for (int d = 0; d < 16; ++d)
        acc += W[(t * 64 + i) * 64 + h * 16 + d] * R[((h * 2 + te) * 16 + d) * 16 + f];
      v = acc;
    }
    Bmat[idx] = v;
  }
  __syncthreads();
  // Phase 2: pack 80 B fragments (2 types x 20 col-tiles x 2 K-halves)
  for (int idx = tid; idx < 80 * 512; idx += 256) {
    int fid = idx >> 9;
    int rem = idx & 511;
    int l   = rem >> 4;   // lane
    int j   = rem & 15;   // element within v16h
    int t   = fid / 40;
    int r2  = fid % 40;
    int ct  = r2 >> 1;
    int kf  = r2 & 1;
    int K   = kf * 32 + ((l & 16) ? 8 : 0) + (j & 7) + ((j & 8) ? 16 : 0);
    int col = ct * 16 + (l & 15);
    fragQKV[idx] = (_Float16)Bmat[(t * 64 + K) * 320 + col];
  }
  // Phase 3: pack 16 Wa fragments (2 types x 4 col-tiles x 2 K-halves)
  for (int idx = tid; idx < 16 * 512; idx += 256) {
    int fid = idx >> 9;
    int rem = idx & 511;
    int l   = rem >> 4;
    int j   = rem & 15;
    int t   = fid / 8;
    int r2  = fid % 8;
    int ct  = r2 >> 1;
    int kf  = r2 & 1;
    int K   = kf * 32 + ((l & 16) ? 8 : 0) + (j & 7) + ((j & 8) ? 16 : 0);
    int col = ct * 16 + (l & 15);
    fragWa[idx] = (_Float16)Wa[(t * 64 + K) * 64 + col];
  }
}

// ---------------------------------------------------------------------------
// Zero accumulators (ws is poisoned; graph replays need fresh zeros each call)
// ---------------------------------------------------------------------------
__global__ void __launch_bounds__(256) hgt_zero(float* __restrict__ asum,
                                                float* __restrict__ hagg,
                                                int n_asum, int n_hagg)
{
  int i      = blockIdx.x * blockDim.x + threadIdx.x;
  int stride = gridDim.x * blockDim.x;
  for (int k = i; k < n_asum; k += stride) asum[k] = 0.f;
  for (int k = i; k < n_hagg; k += stride) hagg[k] = 0.f;
}

// ---------------------------------------------------------------------------
// QKV + rotation GEMM: one wave per 16-node tile. The 80KB B-fragment table
// is async-staged into LDS once per block (shared by all 8 waves), then each
// wave runs 80 v_wmma computing the 16x320 product for BOTH node types and
// selects per row by ntype.
// Writes q[N][64], katt[ET][N][64], vmsg[ET][N][64] (edge-gather friendly).
// ---------------------------------------------------------------------------
__global__ void __launch_bounds__(256) hgt_qkv(
    const float* __restrict__ x, const int* __restrict__ ntype,
    const _Float16* __restrict__ fragQKV,
    float* __restrict__ qbuf, float* __restrict__ katt, float* __restrict__ vmsg,
    int N)
{
  __shared__ __align__(256) _Float16 sfrag[80 * 512];   // 80 KB of WGP LDS

  // Async global -> LDS stage: 256 threads x 20 x 16B = 80KB, ASYNCcnt-tracked.
  {
    const uint32_t lds0 = (uint32_t)(uintptr_t)sfrag;
    #pragma unroll 4
    for (int i = threadIdx.x; i < (80 * 512) / 8; i += 256) {
      async_copy_b128(lds0 + (uint32_t)i * 16,
                      (uint64_t)(uintptr_t)(fragQKV + (size_t)i * 8));
    }
    async_wait0();
  }
  __syncthreads();

  const int lane = threadIdx.x & 31;
  const int wave = threadIdx.x >> 5;
  const int tile = blockIdx.x * 8 + wave;
  const int ntiles = N >> 4;
  if (tile >= ntiles) return;

  const int hi  = lane >> 4;       // lane half (K-chunk select for A)
  const int row = tile * 16 + (lane & 15);
  const float* xr = x + (size_t)row * 64;

  // A fragments (16-bit A 16x32 layout, ISA 7.12.2): two K halves of x row.
  v16h a0{}, a1{};
  #pragma unroll
  for (int j = 0; j < 8; ++j) {
    a0[j]     = (_Float16)xr[hi * 8 + j];
    a0[8 + j] = (_Float16)xr[16 + hi * 8 + j];
    a1[j]     = (_Float16)xr[32 + hi * 8 + j];
    a1[8 + j] = (_Float16)xr[48 + hi * 8 + j];
  }

  // node types for this lane's 8 output rows (C layout: M = r + hi*8)
  const int rowbase = tile * 16 + hi * 8;
  int nt[8];
  #pragma unroll
  for (int r = 0; r < 8; ++r) nt[r] = ntype[rowbase + r];

  const v16h* frags = (const v16h*)sfrag;   // LDS-resident fragments

  #pragma unroll
  for (int ct = 0; ct < 20; ++ct) {
    v16h b00 = frags[((0 * 20 + ct) * 2 + 0) * 32 + lane];
    v16h b01 = frags[((0 * 20 + ct) * 2 + 1) * 32 + lane];
    v16h b10 = frags[((1 * 20 + ct) * 2 + 0) * 32 + lane];
    v16h b11 = frags[((1 * 20 + ct) * 2 + 1) * 32 + lane];
    v8f c0 = {}; v8f c1 = {};
    c0 = WMMA_F16(a0, b00, c0);
    c0 = WMMA_F16(a1, b01, c0);
    c1 = WMMA_F16(a0, b10, c1);
    c1 = WMMA_F16(a1, b11, c1);

    const int grp  = ct >> 2;                   // 0:q 1:katt0 2:katt1 3:vmsg0 4:vmsg1
    const int ccol = (ct & 3) * 16 + (lane & 15);
    float* out = (grp == 0) ? qbuf
               : (grp == 1) ? katt
               : (grp == 2) ? (katt + (size_t)N * 64)
               : (grp == 3) ? vmsg
                            : (vmsg + (size_t)N * 64);
    #pragma unroll
    for (int r = 0; r < 8; ++r) {
      float v = nt[r] ? c1[r] : c0[r];
      out[(size_t)(rowbase + r) * 64 + ccol] = v;
    }
  }
}

// ---------------------------------------------------------------------------
// Edge pass: 16 lanes per edge (4 lanes per head). Since a = O(0.1),
// softmax-max subtraction is mathematically redundant: attn = e^a / sum e^a.
// Accumulate hagg[dst] += vmsg*e^a and asum[dst] += e^a with HW FP32 atomics;
// the final kernel divides. All gathers hit L2 (tables ~64MB < 192MB L2).
// ---------------------------------------------------------------------------
__global__ void __launch_bounds__(256) hgt_edge(
    const float* __restrict__ qbuf, const float* __restrict__ katt,
    const float* __restrict__ vmsg, const float* __restrict__ pri,
    const int* __restrict__ etype, const int* __restrict__ src,
    const int* __restrict__ dst,
    float* __restrict__ asum, float* __restrict__ hagg, int N, int E)
{
  const int gid = blockIdx.x * 256 + threadIdx.x;
  const int e   = gid >> 4;
  if (e >= E) return;
  const int sub  = gid & 15;
  const int head = sub >> 2;
  const int dp   = sub & 3;

  const int et = etype[e];
  const int s  = src[e];
  const int d  = dst[e];

  const size_t kbase = ((size_t)et * N + s) * 64 + head * 16 + dp * 4;
  const float4 kv = *(const float4*)(katt + kbase);
  const float4 qv = *(const float4*)(qbuf + (size_t)d * 64 + head * 16 + dp * 4);

  float p = kv.x * qv.x + kv.y * qv.y + kv.z * qv.z + kv.w * qv.w;
  p += __shfl_xor(p, 1);
  p += __shfl_xor(p, 2);   // full 16-wide dot per head across the 4-lane group

  const float a  = p * pri[head * 2 + et] * 0.25f;   // 1/sqrt(D) = 1/4
  const float ae = __expf(a);

  const float4 mv = *(const float4*)(vmsg + kbase);
  float* hb = hagg + (size_t)d * 64 + head * 16 + dp * 4;
  unsafeAtomicAdd(hb + 0, mv.x * ae);
  unsafeAtomicAdd(hb + 1, mv.y * ae);
  unsafeAtomicAdd(hb + 2, mv.z * ae);
  unsafeAtomicAdd(hb + 3, mv.w * ae);
  if (dp == 0) unsafeAtomicAdd(asum + (size_t)d * 4 + head, ae);
}

// ---------------------------------------------------------------------------
// Final: normalize aggregate, typed linear with Wa (WMMA via LDS-staged
// fragments, both types + per-row select), sigmoid-skip blend with x.
// ---------------------------------------------------------------------------
__global__ void __launch_bounds__(256) hgt_final(
    const float* __restrict__ x, const int* __restrict__ ntype,
    const float* __restrict__ hagg, const float* __restrict__ asum,
    const _Float16* __restrict__ fragWa, const float* __restrict__ skip,
    float* __restrict__ out, int N)
{
  __shared__ __align__(256) _Float16 sfrag[16 * 512];   // 16 KB

  {
    const uint32_t lds0 = (uint32_t)(uintptr_t)sfrag;
    #pragma unroll 4
    for (int i = threadIdx.x; i < (16 * 512) / 8; i += 256) {
      async_copy_b128(lds0 + (uint32_t)i * 16,
                      (uint64_t)(uintptr_t)(fragWa + (size_t)i * 8));
    }
    async_wait0();
  }
  __syncthreads();

  const int lane = threadIdx.x & 31;
  const int wave = threadIdx.x >> 5;
  const int tile = blockIdx.x * 8 + wave;
  if (tile >= (N >> 4)) return;

  const int hi  = lane >> 4;
  const int row = tile * 16 + (lane & 15);

  float inv[4];
  #pragma unroll
  for (int h = 0; h < 4; ++h) {
    float s = asum[(size_t)row * 4 + h];
    inv[h] = (s > 0.f) ? (1.f / s) : 0.f;   // edgeless nodes -> h = 0
  }

  const float* hr = hagg + (size_t)row * 64;
  v16h a0{}, a1{};
  #pragma unroll
  for (int j = 0; j < 8; ++j) {
    a0[j]     = (_Float16)(hr[hi * 8 + j]      * inv[0]);   // cols  0..15 head0
    a0[8 + j] = (_Float16)(hr[16 + hi * 8 + j] * inv[1]);   // cols 16..31 head1
    a1[j]     = (_Float16)(hr[32 + hi * 8 + j] * inv[2]);   // cols 32..47 head2
    a1[8 + j] = (_Float16)(hr[48 + hi * 8 + j] * inv[3]);   // cols 48..63 head3
  }

  const int rowbase = tile * 16 + hi * 8;
  int nt[8];
  #pragma unroll
  for (int r = 0; r < 8; ++r) nt[r] = ntype[rowbase + r];

  const float al0 = 1.f / (1.f + __expf(-skip[0]));
  const float al1 = 1.f / (1.f + __expf(-skip[1]));

  const v16h* frags = (const v16h*)sfrag;
  #pragma unroll
  for (int ct = 0; ct < 4; ++ct) {
    v16h b00 = frags[((0 * 4 + ct) * 2 + 0) * 32 + lane];
    v16h b01 = frags[((0 * 4 + ct) * 2 + 1) * 32 + lane];
    v16h b10 = frags[((1 * 4 + ct) * 2 + 0) * 32 + lane];
    v16h b11 = frags[((1 * 4 + ct) * 2 + 1) * 32 + lane];
    v8f c0 = {}; v8f c1 = {};
    c0 = WMMA_F16(a0, b00, c0);
    c0 = WMMA_F16(a1, b01, c0);
    c1 = WMMA_F16(a0, b10, c1);
    c1 = WMMA_F16(a1, b11, c1);

    const int ccol = ct * 16 + (lane & 15);
    #pragma unroll
    for (int r = 0; r < 8; ++r) {
      const int rw    = rowbase + r;
      const float hv  = nt[r] ? c1[r] : c0[r];
      const float al  = nt[r] ? al1 : al0;
      out[(size_t)rw * 64 + ccol] = hv * al + x[(size_t)rw * 64 + ccol] * (1.f - al);
    }
  }
}

// ---------------------------------------------------------------------------
extern "C" void kernel_launch(void* const* d_in, const int* in_sizes, int n_in,
                              void* d_out, int out_size, void* d_ws, size_t ws_size,
                              hipStream_t stream)
{
  const float* x     = (const float*)d_in[0];
  const float* Wk    = (const float*)d_in[1];
  const float* Wq    = (const float*)d_in[2];
  const float* Wv    = (const float*)d_in[3];
  const float* Wa    = (const float*)d_in[4];
  const float* Ratt  = (const float*)d_in[5];
  const float* Rmsg  = (const float*)d_in[6];
  const float* pri   = (const float*)d_in[7];
  const float* skip  = (const float*)d_in[8];
  const int*   ntype = (const int*)d_in[9];
  const int*   etype = (const int*)d_in[10];
  const int*   src   = (const int*)d_in[11];
  const int*   dst   = (const int*)d_in[12];
  float* out = (float*)d_out;

  const int N = in_sizes[9];
  const int E = in_sizes[10];

  // Workspace carve-up (256B aligned slices)
  char* ws = (char*)d_ws;
  size_t off = 0;
  auto take = [&](size_t bytes) -> char* {
    char* p = ws + off;
    off += (bytes + 255) & ~(size_t)255;
    return p;
  };
  float*     qbuf    = (float*)take((size_t)N * 64 * 4);
  float*     katt    = (float*)take((size_t)2 * N * 64 * 4);
  float*     vmsg    = (float*)take((size_t)2 * N * 64 * 4);
  float*     asum    = (float*)take((size_t)N * 4 * 4);
  float*     hagg    = (float*)take((size_t)N * 64 * 4);
  float*     Bmat    = (float*)take((size_t)2 * 64 * 320 * 4);
  _Float16*  fragQKV = (_Float16*)take((size_t)80 * 512 * 2);
  _Float16*  fragWa  = (_Float16*)take((size_t)16 * 512 * 2);

  hgt_prep<<<1, 256, 0, stream>>>(Wk, Wq, Wv, Wa, Ratt, Rmsg, Bmat, fragQKV, fragWa);
  hgt_zero<<<512, 256, 0, stream>>>(asum, hagg, N * 4, N * 64);

  const int ntiles = N / 16;
  const int agrid  = (ntiles + 7) / 8;
  hgt_qkv<<<agrid, 256, 0, stream>>>(x, ntype, fragQKV, qbuf, katt, vmsg, N);

  const int egrid = (E * 16 + 255) / 256;
  hgt_edge<<<egrid, 256, 0, stream>>>(qbuf, katt, vmsg, pri, etype, src, dst,
                                      asum, hagg, N, E);

  hgt_final<<<agrid, 256, 0, stream>>>(x, ntype, hagg, asum, fragWa, skip, out, N);
}
